// DyRep_1614907703404
// MI455X (gfx1250) — matrix operand order repaired and checked by reading, not compiled, not run on hardware
//
#include <hip/hip_runtime.h>
#include <hip/hip_bf16.h>
#include <math.h>

// ---------------------------------------------------------------------------
// DyRep forward for MI455X (gfx1250, wave32).
//
// Sizes: N=8192 nodes, H=32, B=256 events -> P=512 (event,side) pairs.
// Outputs: Lambda[256] then z_new[256*2*32], concatenated in d_out.
//
// Roofline: dominated by scanning A rows (mask) for 512 pairs: 512*32KB=16MB;
// S is read sparsely only where A>0 (~32/8192 elements). ~0.7us at 23.3TB/s.
// h_all GEMM ([8192,32]x[32,32]) done with v_wmma_f32_16x16x32_f16 using a
// split-f16 (hi/lo) 3-WMMA product for near-f32 accuracy.
// ---------------------------------------------------------------------------

typedef __attribute__((ext_vector_type(16))) _Float16 v16h;
typedef __attribute__((ext_vector_type(8)))  float    v8f;

#define NN 8192
#define HH 32
#define BB 256
#define NP 512            // B*2 pairs
#define CAP 1024          // neighbor-list capacity (deg ~ Binom(8192, 32/8192), mean 32)

// ---------------------------------------------------------------------------
// Kernel 1: h_all = z @ W_h_w^T + W_h_b   ([8192,32] f32), one wave per
// 16x16 output tile, K=32 in a single WMMA step (x3 for split precision).
// Fragment layouts per CDNA5 ISA 7.12.2 (wave32):
//   A (16x32 f16): lane m=l&15 is row; VGPR v holds K = 2v(+8 if v>=4) + 8*(l>>4)
//   B (32x16 f16): lane n=l&15 is col; VGPR v holds K = 2v(+8 if v>=4) + 8*(l>>4)
//   C/D (16x16 f32): row = r + 8*(l>>4), col = l&15
// ---------------------------------------------------------------------------
__global__ __launch_bounds__(32)
void k_h_all(const float* __restrict__ z, const float* __restrict__ Ww,
             const float* __restrict__ Wb, float* __restrict__ h_all) {
  const int lane = threadIdx.x;
  const int m  = lane & 15;
  const int hi = lane >> 4;
  const int row0 = blockIdx.x * 16;   // z-row tile
  const int n0   = blockIdx.y * 16;   // output-col tile (0 or 16)

  v16h a_hi, a_lo, b_hi, b_lo;

  // A fragment from z[row0+m, :]
#pragma unroll
  for (int v = 0; v < 8; ++v) {
    const int kb = 2 * v + (v >= 4 ? 8 : 0) + 8 * hi;
#pragma unroll
    for (int e = 0; e < 2; ++e) {
      float x = z[(size_t)(row0 + m) * HH + kb + e];
      _Float16 xh = (_Float16)x;
      a_hi[2 * v + e] = xh;
      a_lo[2 * v + e] = (_Float16)(x - (float)xh);
    }
  }
  // B fragment: B[k,n] = Ww[(n0+n)*32 + k]  (h_all[r,n] = sum_k z[r,k]*Ww[n,k])
#pragma unroll
  for (int v = 0; v < 8; ++v) {
#pragma unroll
    for (int e = 0; e < 2; ++e) {
      const int kk = 2 * v + (v >= 4 ? 8 : 0) + 8 * hi + e;
      float x = Ww[(size_t)(n0 + m) * HH + kk];
      _Float16 xh = (_Float16)x;
      b_hi[2 * v + e] = xh;
      b_lo[2 * v + e] = (_Float16)(x - (float)xh);
    }
  }

  v8f c = {};
  // x*y ~= lo*hi + hi*lo + hi*hi (f32 accumulate); lo*lo term ~2^-22, dropped.
  c = __builtin_amdgcn_wmma_f32_16x16x32_f16(false, a_lo, false, b_hi, (short)0, c, false, false);
  c = __builtin_amdgcn_wmma_f32_16x16x32_f16(false, a_hi, false, b_lo, (short)0, c, false, false);
  c = __builtin_amdgcn_wmma_f32_16x16x32_f16(false, a_hi, false, b_hi, (short)0, c, false, false);

  const float bias = Wb[n0 + m];
#pragma unroll
  for (int r = 0; r < 8; ++r)
    h_all[(size_t)(row0 + r + 8 * hi) * HH + (n0 + m)] = c[r] + bias;
}

// ---------------------------------------------------------------------------
// Kernel 2: Lambda[b].  g_fn(uv)+g_fn(vu) collapses to dot(zu+zv, w0+w1)+2b.
// ---------------------------------------------------------------------------
__global__ void k_lambda(const float* __restrict__ z, const int* __restrict__ u_idx,
                         const int* __restrict__ v_idx, const int* __restrict__ kev,
                         const float* __restrict__ om0w, const float* __restrict__ om0b,
                         const float* __restrict__ om1w, const float* __restrict__ om1b,
                         const float* __restrict__ psi, float* __restrict__ out) {
  const int b = blockIdx.x * blockDim.x + threadIdx.x;
  if (b >= BB) return;
  const bool kpos = kev[b] > 0;
  const float* w = kpos ? om1w : om0w;
  const float bias = kpos ? om1b[0] : om0b[0];
  const float* zu = z + (size_t)u_idx[b] * HH;
  const float* zv = z + (size_t)v_idx[b] * HH;
  float acc = 0.f;
#pragma unroll
  for (int h = 0; h < HH; ++h)
    acc += (zu[h] + zv[h]) * (w[h] + w[HH + h]);
  const float g  = 0.5f * acc + bias;
  const float ps = psi[kpos ? 1 : 0];
  float gp = g / (ps + 1e-7f);
  gp = fminf(fmaxf(gp, -75.f), 75.f);
  out[b] = ps * (log1pf(expf(-gp)) + gp);
}

// ---------------------------------------------------------------------------
// Kernel 3: per-pair neighbor scan + masked max.  One 256-thread block per
// pair P (b=P>>1, side j=P&1; partner = j==0 ? v : u).
// Phase 1: stream A row (float4, coalesced), sparse-read S only where A>0,
//          compact (n, exp(S)) into LDS; sum q with a deterministic tree.
// Phase 2: h_struct[h] = sigmoid( max_i q_i*h_all[n_i,h] / (Sum q + 1e-7) )
//          (monotone sigmoid + positive common scale => exact transform).
// ---------------------------------------------------------------------------
__global__ __launch_bounds__(256)
void k_scan(const float* __restrict__ A, const float* __restrict__ S,
            const float* __restrict__ h_all, const int* __restrict__ u_idx,
            const int* __restrict__ v_idx, float* __restrict__ h_struct) {
  __shared__ int   s_idx[CAP];
  __shared__ float s_q[CAP];
  __shared__ float s_red[256];
  __shared__ float s_part[8][32];
  __shared__ int   s_cnt;

  const int t = threadIdx.x;
  const int P = blockIdx.x;
  const int b = P >> 1, j = P & 1;
  const int p = (j == 0) ? v_idx[b] : u_idx[b];

  if (t == 0) s_cnt = 0;
  __syncthreads();

  const float* Arow = A + (size_t)p * NN;
  const float* Srow = S + (size_t)p * NN;

  float qloc = 0.f;
  for (int base = t * 4; base < NN; base += 256 * 4) {
    const float4 av = *(const float4*)(Arow + base);
    const float aarr[4] = {av.x, av.y, av.z, av.w};
#pragma unroll
    for (int e = 0; e < 4; ++e) {
      if (aarr[e] > 0.f) {                 // sparse: ~32 of 8192 hit
        const float q = expf(Srow[base + e]);
        const int slot = atomicAdd(&s_cnt, 1);
        if (slot < CAP) { s_idx[slot] = base + e; s_q[slot] = q; }
        qloc += q;
      }
    }
  }
  // deterministic tree-sum of q (no float atomics)
  s_red[t] = qloc;
  __syncthreads();
  for (int off = 128; off > 0; off >>= 1) {
    if (t < off) s_red[t] += s_red[t + off];
    __syncthreads();
  }
  const float Qsum = s_red[0];
  const int cnt = min(s_cnt, CAP);

  // Phase 2: 8 groups x 32 columns; fmax is order-independent -> deterministic.
  const int h = t & 31, g = t >> 5;
  float M = -INFINITY;
  for (int i = g; i < cnt; i += 8)
    M = fmaxf(M, s_q[i] * h_all[(size_t)s_idx[i] * HH + h]);
  s_part[g][h] = M;
  __syncthreads();

  if (t < 32) {
    float mm = s_part[0][t];
#pragma unroll
    for (int gg = 1; gg < 8; ++gg) mm = fmaxf(mm, s_part[gg][t]);
    float res = 0.f;
    if (cnt > 0) {
      const float x = mm / (Qsum + 1e-7f);
      res = 1.f / (1.f + expf(-x));
    }
    h_struct[(size_t)P * HH + t] = res;
  }
}

// ---------------------------------------------------------------------------
// Kernel 4: z_new = sigmoid(W_s h_struct + b_s + W_r z[target] + b_r
//                           + W_t td + b_t).  512x32 outputs, exact f32.
// ---------------------------------------------------------------------------
__global__ __launch_bounds__(256)
void k_znew(const float* __restrict__ z, const int* __restrict__ u_idx,
            const int* __restrict__ v_idx, const float* __restrict__ td,
            const float* __restrict__ hs,
            const float* __restrict__ Wsw, const float* __restrict__ Wsb,
            const float* __restrict__ Wrw, const float* __restrict__ Wrb,
            const float* __restrict__ Wtw, const float* __restrict__ Wtb,
            float* __restrict__ out) {
  const int idx = blockIdx.x * blockDim.x + threadIdx.x;
  if (idx >= NP * HH) return;
  const int h = idx & 31;
  const int P = idx >> 5;
  const int b = P >> 1, j = P & 1;
  const int tgt = (j == 0) ? u_idx[b] : v_idx[b];

  const float* hrow  = hs + (size_t)P * HH;
  const float* zrow  = z + (size_t)tgt * HH;
  const float* tdrow = td + (size_t)P * 4;

  float acc = Wsb[h] + Wrb[h] + Wtb[h];
#pragma unroll
  for (int k2 = 0; k2 < HH; ++k2)
    acc += Wsw[h * HH + k2] * hrow[k2] + Wrw[h * HH + k2] * zrow[k2];
#pragma unroll
  for (int k2 = 0; k2 < 4; ++k2)
    acc += Wtw[h * 4 + k2] * tdrow[k2];

  out[BB + idx] = 1.f / (1.f + expf(-acc));
}

// ---------------------------------------------------------------------------
extern "C" void kernel_launch(void* const* d_in, const int* in_sizes, int n_in,
                              void* d_out, int out_size, void* d_ws, size_t ws_size,
                              hipStream_t stream) {
  const float* z    = (const float*)d_in[0];
  const float* A    = (const float*)d_in[1];
  const float* S    = (const float*)d_in[2];
  const int*   u_i  = (const int*)d_in[3];
  const int*   v_i  = (const int*)d_in[4];
  const int*   kev  = (const int*)d_in[5];
  const float* td   = (const float*)d_in[6];
  const float* Whw  = (const float*)d_in[7];
  const float* Whb  = (const float*)d_in[8];
  const float* Wsw  = (const float*)d_in[9];
  const float* Wsb  = (const float*)d_in[10];
  const float* Wrw  = (const float*)d_in[11];
  const float* Wrb  = (const float*)d_in[12];
  const float* Wtw  = (const float*)d_in[13];
  const float* Wtb  = (const float*)d_in[14];
  const float* om0w = (const float*)d_in[15];
  const float* om0b = (const float*)d_in[16];
  const float* om1w = (const float*)d_in[17];
  const float* om1b = (const float*)d_in[18];
  const float* psi  = (const float*)d_in[19];

  float* out      = (float*)d_out;
  float* h_all    = (float*)d_ws;                 // 8192*32 f32 = 1 MB
  float* h_struct = h_all + (size_t)NN * HH;      // 512*32 f32

  // 1) h_all via WMMA (512 row-tiles x 2 col-tiles, one wave each)
  k_h_all<<<dim3(NN / 16, HH / 16), 32, 0, stream>>>(z, Whw, Whb, h_all);
  // 2) Lambda (independent of h_all)
  k_lambda<<<1, BB, 0, stream>>>(z, u_i, v_i, kev, om0w, om0b, om1w, om1b, psi, out);
  // 3) neighbor scan + masked max (the bandwidth-bound stage)
  k_scan<<<NP, 256, 0, stream>>>(A, S, h_all, u_i, v_i, h_struct);
  // 4) final update
  k_znew<<<(NP * HH + 255) / 256, 256, 0, stream>>>(z, u_i, v_i, td, h_struct,
                                                    Wsw, Wsb, Wrw, Wrb, Wtw, Wtb, out);
}
